// KLContrastivePULoss_6270652252920
// MI455X (gfx1250) — compile-verified, matrix operand order checked by reading, not run on hardware
//
#include <hip/hip_runtime.h>
#include <hip/hip_bf16.h>

// ---------------------------------------------------------------------------
// KLContrastivePULoss for MI455X (gfx1250, wave32, WMMA)
// Dominant cost: 8192x8192x128 GEMM (17.2 GFLOP) + exp + row-sum, fused.
// fp16 WMMA (V_WMMA_F32_16X16X32_F16); B tiles staged through LDS (8x reuse),
// 2 M-tiles register-blocked per wave (2x reuse) -> ~128 MB L2 traffic total.
// ---------------------------------------------------------------------------

typedef _Float16 v16h __attribute__((ext_vector_type(16)));
typedef _Float16 v8h  __attribute__((ext_vector_type(8)));
typedef _Float16 v4h  __attribute__((ext_vector_type(4)));
typedef float    v8f  __attribute__((ext_vector_type(8)));

#define N_ROWS 8192
#define DIM    128
// exp(x / 0.5) == exp2(x * 2 * log2(e))
#define EXP_SCALE 2.8853900817779268f

// LDS B-chunk: 128 columns (z_j rows) x 128 K halves, padded row pitch.
// Pitch 136 halves = 272 B = 68 banks -> row r starts at bank (4*r)%64, so the
// 16 lanes of a ds_load_b128 touch 64 distinct banks. 34 KB of 320 KB LDS.
#define CHUNK_COLS 128
#define ROW_PITCH  136

// ---- wave32 butterfly reductions (all lanes end with the result) ----------
__device__ __forceinline__ float wred_sum(float v) {
#pragma unroll
  for (int off = 16; off > 0; off >>= 1) v += __shfl_xor(v, off, 32);
  return v;
}
__device__ __forceinline__ float wred_max(float v) {
#pragma unroll
  for (int off = 16; off > 0; off >>= 1) v = fmaxf(v, __shfl_xor(v, off, 32));
  return v;
}

// ---------------------------------------------------------------------------
// Kernel 1: per-row prep. One wave per row, 4 floats per lane.
//  - L2 normalize z_i, z_j -> fp16 copies (row-major [N][128])
//  - pos_dot[n] = <zi_n, zj_n>  (fp32)
//  - row_kl[n]  = sum_d q*(logq - logp) with stable log-softmax
//  - neg_sim[n] = 0  (accumulator init; harness does not re-zero ws)
// ---------------------------------------------------------------------------
__global__ void __launch_bounds__(256)
prep_kernel(const float* __restrict__ zi, const float* __restrict__ zj,
            _Float16* __restrict__ zi_h, _Float16* __restrict__ zj_h,
            float* __restrict__ neg_sim, float* __restrict__ pos_dot,
            float* __restrict__ row_kl) {
  const int lane = threadIdx.x & 31;
  const int wave = threadIdx.x >> 5;
  const int row  = blockIdx.x * 8 + wave;
  if (row >= N_ROWS) return;

  const float4 xi = ((const float4*)(zi + (size_t)row * DIM))[lane];
  const float4 xj = ((const float4*)(zj + (size_t)row * DIM))[lane];

  // --- L2 norms (eps = 1e-8 as in torch CosineSimilarity) ---
  const float ni2 = wred_sum(xi.x*xi.x + xi.y*xi.y + xi.z*xi.z + xi.w*xi.w);
  const float nj2 = wred_sum(xj.x*xj.x + xj.y*xj.y + xj.z*xj.z + xj.w*xj.w);
  const float inv_i = 1.0f / fmaxf(sqrtf(ni2), 1e-8f);
  const float inv_j = 1.0f / fmaxf(sqrtf(nj2), 1e-8f);

  const float ax = xi.x*inv_i, ay = xi.y*inv_i, az = xi.z*inv_i, aw = xi.w*inv_i;
  const float bx = xj.x*inv_j, by = xj.y*inv_j, bz = xj.z*inv_j, bw = xj.w*inv_j;

  // --- positive-pair dot in fp32 ---
  const float pos = wred_sum(ax*bx + ay*by + az*bz + aw*bw);

  // --- fp16 normalized copies for the WMMA GEMM ---
  v4h ha = { (_Float16)ax, (_Float16)ay, (_Float16)az, (_Float16)aw };
  v4h hb = { (_Float16)bx, (_Float16)by, (_Float16)bz, (_Float16)bw };
  ((v4h*)(zi_h + (size_t)row * DIM))[lane] = ha;
  ((v4h*)(zj_h + (size_t)row * DIM))[lane] = hb;

  // --- stable log-softmax stats for KL ---
  const float mi = wred_max(fmaxf(fmaxf(xi.x, xi.y), fmaxf(xi.z, xi.w)));
  const float mj = wred_max(fmaxf(fmaxf(xj.x, xj.y), fmaxf(xj.z, xj.w)));
  const float si = wred_sum(expf(xi.x-mi) + expf(xi.y-mi) + expf(xi.z-mi) + expf(xi.w-mi));
  const float sj = wred_sum(expf(xj.x-mj) + expf(xj.y-mj) + expf(xj.z-mj) + expf(xj.w-mj));
  const float lse_i = mi + logf(si);
  const float lse_j = mj + logf(sj);

  float kl = 0.0f;
  {
    float lq, lp;
    lq = xj.x - lse_j; lp = xi.x - lse_i; kl += expf(lq) * (lq - lp);
    lq = xj.y - lse_j; lp = xi.y - lse_i; kl += expf(lq) * (lq - lp);
    lq = xj.z - lse_j; lp = xi.z - lse_i; kl += expf(lq) * (lq - lp);
    lq = xj.w - lse_j; lp = xi.w - lse_i; kl += expf(lq) * (lq - lp);
  }
  kl = wred_sum(kl);

  if (lane == 0) {
    pos_dot[row] = pos;
    row_kl[row]  = kl;
    neg_sim[row] = 0.0f;
  }
}

// ---------------------------------------------------------------------------
// Per-lane v16h operand for the 16-bit 16x32 WMMA layout:
//   lanes 0-15 : K = kb*32 + {0..7, 16..23}
//   lanes 16-31: K = kb*32 + {8..15, 24..31}
// rowp points at one tile-row (global fp16 row or LDS padded row);
// two 16B loads per K-block. Inlined so addrspace inference keeps ds_load.
// ---------------------------------------------------------------------------
__device__ __forceinline__ v16h load_k(const _Float16* rowp, int kb, int khalf) {
  const v8h lo = *(const v8h*)(rowp + kb * 32 + khalf * 8);
  const v8h hi = *(const v8h*)(rowp + kb * 32 + 16 + khalf * 8);
  return __builtin_shufflevector(lo, hi, 0,1,2,3,4,5,6,7,8,9,10,11,12,13,14,15);
}

#define WMMA_F16(A, B, C) \
  __builtin_amdgcn_wmma_f32_16x16x32_f16(false, (A), false, (B), (short)0, (C), \
                                         false, false)

// ---------------------------------------------------------------------------
// Kernel 2: fused sim = A*B^T, exp(sim/T), row-sum.
// Block = 256 threads (8 waves). Each wave owns TWO 16-row M-tiles (A resident
// in 64 VGPRs). Block sweeps 512 columns in 4 LDS-staged chunks of 128; all 8
// waves read the shared B chunk from LDS. Grid (32, 16) = 512 blocks.
// ---------------------------------------------------------------------------
__global__ void __launch_bounds__(256)
gemm_exp_rowsum_kernel(const _Float16* __restrict__ Ah,
                       const _Float16* __restrict__ Bh,
                       float* __restrict__ neg_sim) {
  __shared__ _Float16 sB[CHUNK_COLS * ROW_PITCH];   // 34,816 B

  const int lane  = threadIdx.x & 31;
  const int wave  = threadIdx.x >> 5;
  const int khalf = lane >> 4;     // 0: K-low half, 1: K-high half
  const int r     = lane & 15;     // row-in-tile / col-in-tile

  const int mt0    = blockIdx.x * 16 + wave * 2;    // two M-tiles per wave
  const int n_base = blockIdx.y * 512;              // 16 splits of 512 cols

  // A tiles: 2 x (16 rows x 128 K), resident for the whole sweep (64 VGPRs).
  const _Float16* arow0 = Ah + (size_t)(mt0 * 16 + r) * DIM;
  const _Float16* arow1 = arow0 + 16 * DIM;
  const v16h a00 = load_k(arow0, 0, khalf);
  const v16h a01 = load_k(arow0, 1, khalf);
  const v16h a02 = load_k(arow0, 2, khalf);
  const v16h a03 = load_k(arow0, 3, khalf);
  const v16h a10 = load_k(arow1, 0, khalf);
  const v16h a11 = load_k(arow1, 1, khalf);
  const v16h a12 = load_k(arow1, 2, khalf);
  const v16h a13 = load_k(arow1, 3, khalf);

  float acc0[8] = {0.f, 0.f, 0.f, 0.f, 0.f, 0.f, 0.f, 0.f};
  float acc1[8] = {0.f, 0.f, 0.f, 0.f, 0.f, 0.f, 0.f, 0.f};

  for (int chunk = 0; chunk < 4; ++chunk) {
    // ---- cooperative stage: 128 z_j rows x 128 halves -> padded LDS ----
    __syncthreads();   // previous chunk's readers are done
#pragma unroll
    for (int it = 0; it < 8; ++it) {
      const int p   = threadIdx.x + it * 256;       // 0..2047 b128 pieces
      const int row = p >> 4;
      const int seg = p & 15;
      const v8h d = *(const v8h*)(Bh +
          (size_t)(n_base + chunk * CHUNK_COLS + row) * DIM + seg * 8);
      *(v8h*)(sB + row * ROW_PITCH + seg * 8) = d;
    }
    __syncthreads();

    // ---- compute: 8 N-tiles per chunk, B operands from LDS ----
    for (int nt = 0; nt < 8; ++nt) {
      const _Float16* brow = sB + (nt * 16 + r) * ROW_PITCH;
      const v16h b0 = load_k(brow, 0, khalf);
      const v16h b1 = load_k(brow, 1, khalf);
      const v16h b2 = load_k(brow, 2, khalf);
      const v16h b3 = load_k(brow, 3, khalf);

      v8f c0 = {};
      c0 = WMMA_F16(a00, b0, c0);
      c0 = WMMA_F16(a01, b1, c0);
      c0 = WMMA_F16(a02, b2, c0);
      c0 = WMMA_F16(a03, b3, c0);
      v8f c1 = {};
      c1 = WMMA_F16(a10, b0, c1);
      c1 = WMMA_F16(a11, b1, c1);
      c1 = WMMA_F16(a12, b2, c1);
      c1 = WMMA_F16(a13, b3, c1);

      // C/D layout: lane 0-15 -> (M=v, N=lane); lane 16-31 -> (M=v+8, N=lane-16)
#pragma unroll
      for (int v = 0; v < 8; ++v) {
        acc0[v] += __builtin_amdgcn_exp2f(c0[v] * EXP_SCALE);
        acc1[v] += __builtin_amdgcn_exp2f(c1[v] * EXP_SCALE);
      }
    }
  }

  // Collapse N within each 16-lane half: lanes {0,16} hold the 8 row sums.
#pragma unroll
  for (int v = 0; v < 8; ++v) {
    float s0 = acc0[v], s1 = acc1[v];
    s0 += __shfl_xor(s0, 1, 32);  s1 += __shfl_xor(s1, 1, 32);
    s0 += __shfl_xor(s0, 2, 32);  s1 += __shfl_xor(s1, 2, 32);
    s0 += __shfl_xor(s0, 4, 32);  s1 += __shfl_xor(s1, 4, 32);
    s0 += __shfl_xor(s0, 8, 32);  s1 += __shfl_xor(s1, 8, 32);
    acc0[v] = s0; acc1[v] = s1;
  }
  if (r == 0) {
    const int row0 = mt0 * 16 + khalf * 8;
#pragma unroll
    for (int v = 0; v < 8; ++v) {
      atomicAdd(neg_sim + row0 + v,      acc0[v]);
      atomicAdd(neg_sim + row0 + 16 + v, acc1[v]);
    }
  }
}

// ---------------------------------------------------------------------------
// Kernel 3: final scalar. log(pos_sim) == pos_dot/T exactly, so
//   contrastive[n] = log(sum_exp + 1e-9) - 2*pos_dot[n]
//   out = (sum_n uw[n]*contrastive[n] + sum_n row_kl[n]) / N
// ---------------------------------------------------------------------------
__global__ void __launch_bounds__(256)
final_reduce_kernel(const float* __restrict__ neg_sim,
                    const float* __restrict__ pos_dot,
                    const float* __restrict__ row_kl,
                    const float* __restrict__ uw,
                    float* __restrict__ out) {
  __shared__ float sdata[256];
  float local = 0.0f;
  for (int n = threadIdx.x; n < N_ROWS; n += 256) {
    const float ct = logf(neg_sim[n] + 1e-9f) - 2.0f * pos_dot[n];
    local += uw[n] * ct + row_kl[n];
  }
  sdata[threadIdx.x] = local;
  __syncthreads();
#pragma unroll
  for (int s = 128; s > 0; s >>= 1) {
    if (threadIdx.x < s) sdata[threadIdx.x] += sdata[threadIdx.x + s];
    __syncthreads();
  }
  if (threadIdx.x == 0) out[0] = sdata[0] / (float)N_ROWS;
}

// ---------------------------------------------------------------------------
extern "C" void kernel_launch(void* const* d_in, const int* in_sizes, int n_in,
                              void* d_out, int out_size, void* d_ws, size_t ws_size,
                              hipStream_t stream) {
  const float* zi = (const float*)d_in[0];
  const float* zj = (const float*)d_in[1];
  // d_in[2] = label_indicator (unused by the reference computation)
  const float* uw = (const float*)d_in[3];

  char* ws = (char*)d_ws;
  _Float16* zi_h   = (_Float16*)(ws);                       // 2 MB
  _Float16* zj_h   = (_Float16*)(ws + (size_t)2097152);     // 2 MB
  float*    negsim = (float*)(ws + (size_t)4194304);        // 32 KB
  float*    posdot = (float*)(ws + (size_t)4227072);        // 32 KB
  float*    rowkl  = (float*)(ws + (size_t)4259840);        // 32 KB

  prep_kernel<<<N_ROWS / 8, 256, 0, stream>>>(zi, zj, zi_h, zj_h,
                                              negsim, posdot, rowkl);
  gemm_exp_rowsum_kernel<<<dim3(32, 16), 256, 0, stream>>>(zi_h, zj_h, negsim);
  final_reduce_kernel<<<1, 256, 0, stream>>>(negsim, posdot, rowkl, uw,
                                             (float*)d_out);
}
